// LocalSoftmaxAttention_32719060861270
// MI455X (gfx1250) — matrix operand order repaired and checked by reading, not compiled, not run on hardware
//
#include <hip/hip_runtime.h>
#include <hip/hip_bf16.h>

typedef __attribute__((ext_vector_type(16))) __bf16 v16bf;
typedef __attribute__((ext_vector_type(8)))  float  v8f;

static __device__ __forceinline__ void pack8(v16bf& A, int off, float4 u, float4 w) {
    A[off+0] = (__bf16)u.x; A[off+1] = (__bf16)u.y;
    A[off+2] = (__bf16)u.z; A[off+3] = (__bf16)u.w;
    A[off+4] = (__bf16)w.x; A[off+5] = (__bf16)w.y;
    A[off+6] = (__bf16)w.z; A[off+7] = (__bf16)w.w;
}

// ---------------------------------------------------------------------------
// Kernel 1: fused QKV projection.  q/k/v[row][col] = x[row][:]·W[col][:] + b
// One wave computes one 16x16 tile for all three of q,k,v (shared A operand).
// Block = 256 threads = 8 waves = 8 column tiles (covers all H=128 features).
// Grid  = (B*N)/16 = 2048 token tiles.
// ---------------------------------------------------------------------------
__global__ __launch_bounds__(256) void qkv_wmma_kernel(
    const float* __restrict__ x,
    const float* __restrict__ Wq, const float* __restrict__ bq,
    const float* __restrict__ Wk, const float* __restrict__ bk,
    const float* __restrict__ Wv, const float* __restrict__ bv,
    float* __restrict__ qo, float* __restrict__ ko, float* __restrict__ vo)
{
    const int lane = threadIdx.x & 31;
    const int wave = threadIdx.x >> 5;          // column tile 0..7
    const int tile = blockIdx.x;                // 16 token rows
    const int m    = lane & 15;
    const int hi   = lane >> 4;                 // upper half-wave flag
    const int rowA = tile * 16 + m;             // token row for A operand
    const int col  = wave * 16 + m;             // output feature for B/C/D

    v8f accq = {}; v8f acck = {}; v8f accv = {};

#pragma unroll
    for (int kk = 0; kk < 128; kk += 32) {
        // ---- A: x tile, 16x32 bf16 layout (row m; contiguous K kb..kb+15)
        const int kb = kk + 16 * hi;
        const float4* ap = (const float4*)(x + (size_t)rowA * 128 + kb);
        float4 a0 = ap[0], a1 = ap[1];
        const float4* ap2 = (const float4*)(x + (size_t)rowA * 128 + kb + 8);
        float4 a2 = ap2[0], a3 = ap2[1];
        v16bf A;
        pack8(A, 0, a0, a1);
        pack8(A, 8, a2, a3);

        // ---- B: W[col][kB .. kB+15] (32x16 layout: contiguous 16 K per lane)
        const int kB = kk + 16 * hi;
        v16bf Bq, Bk, Bv;
        {
            const float4* wp = (const float4*)(Wq + (size_t)col * 128 + kB);
            pack8(Bq, 0, wp[0], wp[1]); pack8(Bq, 8, wp[2], wp[3]);
        }
        {
            const float4* wp = (const float4*)(Wk + (size_t)col * 128 + kB);
            pack8(Bk, 0, wp[0], wp[1]); pack8(Bk, 8, wp[2], wp[3]);
        }
        {
            const float4* wp = (const float4*)(Wv + (size_t)col * 128 + kB);
            pack8(Bv, 0, wp[0], wp[1]); pack8(Bv, 8, wp[2], wp[3]);
        }

        accq = __builtin_amdgcn_wmma_f32_16x16x32_bf16(false, A, false, Bq,
                                                       (short)0, accq, false, false);
        acck = __builtin_amdgcn_wmma_f32_16x16x32_bf16(false, A, false, Bk,
                                                       (short)0, acck, false, false);
        accv = __builtin_amdgcn_wmma_f32_16x16x32_bf16(false, A, false, Bv,
                                                       (short)0, accv, false, false);
    }

    const float biasq = bq[col];
    const float biask = bk[col];
    const float biasv = bv[col];
    const int rbase = tile * 16 + 8 * hi;       // D layout: VGPR r -> row r+8*hi
#pragma unroll
    for (int r = 0; r < 8; ++r) {
        const size_t o = (size_t)(rbase + r) * 128 + col;
        qo[o] = accq[r] + biasq;
        ko[o] = acck[r] + biask;
        vo[o] = accv[r] + biasv;
    }
}

// ---------------------------------------------------------------------------
// Kernel 2: gathered local attention, one wave32 per token (K = 32 = wave32).
// Lane l owns channels 4l..4l+3 -> all gathers are fully coalesced 512B rows.
// ---------------------------------------------------------------------------
__global__ __launch_bounds__(256) void attn_kernel(
    const float* __restrict__ q, const float* __restrict__ k,
    const float* __restrict__ v, const int* __restrict__ nbr,
    float* __restrict__ out)
{
    __shared__ int   idx_s[8][32];
    __shared__ float sc_s[8][32 * 9];   // stride 9 kills LDS bank conflicts
    __shared__ float wt_s[8][32 * 9];

    const int lane = threadIdx.x & 31;
    const int wave = threadIdx.x >> 5;
    const int t = blockIdx.x * 8 + wave;        // flat token 0..32767
    const int b = t >> 14;                      // N = 16384
    const int n = t & 16383;
    const int h = lane >> 2;                    // head owned in phases 2/4

    // Phase 1: neighbor indices (coalesced), own q channels.
    idx_s[wave][lane] = (b << 14) + nbr[(size_t)n * 32 + lane];
    const float4 qv = *(const float4*)(q + (size_t)t * 128 + lane * 4);

    // Phase 2: scores s[j][h] with coalesced k-row gathers.
#pragma unroll 4
    for (int j = 0; j < 32; ++j) {
        const int r = idx_s[wave][j];
        const float4 kv = *(const float4*)(k + (size_t)r * 128 + lane * 4);
        float p = kv.x * qv.x + kv.y * qv.y + kv.z * qv.z + kv.w * qv.w;
        p += __shfl_xor(p, 1);
        p += __shfl_xor(p, 2);                  // full d=16 dot in lanes 4h..4h+3
        if ((lane & 3) == 0) sc_s[wave][j * 9 + h] = p * 0.25f;  // 1/sqrt(16)
    }

    // Phase 3: softmax over neighbors (= lane dimension), per head.
#pragma unroll
    for (int hh = 0; hh < 8; ++hh) {
        const float s = sc_s[wave][lane * 9 + hh];
        float m = s;
        m = fmaxf(m, __shfl_xor(m, 16));
        m = fmaxf(m, __shfl_xor(m, 8));
        m = fmaxf(m, __shfl_xor(m, 4));
        m = fmaxf(m, __shfl_xor(m, 2));
        m = fmaxf(m, __shfl_xor(m, 1));
        const float e = __expf(s - m);
        float sum = e;
        sum += __shfl_xor(sum, 16);
        sum += __shfl_xor(sum, 8);
        sum += __shfl_xor(sum, 4);
        sum += __shfl_xor(sum, 2);
        sum += __shfl_xor(sum, 1);
        wt_s[wave][lane * 9 + hh] = e / sum;
    }

    // Phase 4: weighted V with coalesced v-row gathers.
    float4 acc = {0.f, 0.f, 0.f, 0.f};
#pragma unroll 4
    for (int j = 0; j < 32; ++j) {
        const int r = idx_s[wave][j];
        const float w = wt_s[wave][j * 9 + h];
        const float4 vv = *(const float4*)(v + (size_t)r * 128 + lane * 4);
        acc.x += w * vv.x; acc.y += w * vv.y;
        acc.z += w * vv.z; acc.w += w * vv.w;
    }
    *(float4*)(out + (size_t)t * 128 + lane * 4) = acc;
}

// ---------------------------------------------------------------------------
extern "C" void kernel_launch(void* const* d_in, const int* in_sizes, int n_in,
                              void* d_out, int out_size, void* d_ws, size_t ws_size,
                              hipStream_t stream) {
    const float* x  = (const float*)d_in[0];
    const float* Wq = (const float*)d_in[1];
    const float* bq = (const float*)d_in[2];
    const float* Wk = (const float*)d_in[3];
    const float* bk = (const float*)d_in[4];
    const float* Wv = (const float*)d_in[5];
    const float* bv = (const float*)d_in[6];
    const int*  nbr = (const int*)d_in[7];
    float* out = (float*)d_out;

    const int ROWS = 2 * 16384;                 // B*N flattened tokens
    float* qb = (float*)d_ws;                   // 16 MB each
    float* kb = qb + (size_t)ROWS * 128;
    float* vb = kb + (size_t)ROWS * 128;

    qkv_wmma_kernel<<<ROWS / 16, 256, 0, stream>>>(x, Wq, bq, Wk, bk, Wv, bv,
                                                   qb, kb, vb);
    attn_kernel<<<ROWS / 8, 256, 0, stream>>>(qb, kb, vb, nbr, out);
}
